// HHP_6064493822291
// MI455X (gfx1250) — matrix operand order repaired and checked by reading, not compiled, not run on hardware
//
#include <hip/hip_runtime.h>
#include <hip/hip_bf16.h>

#define BSZ   1024
#define NEGN  5
#define NBRN  50
#define DIM   128
#define ET    4
#define NEG_INF (-4294967295.0f)
#define SLOPE 0.01f

// workspace layout (float units)
#define NLAT     (2*BSZ + 2*BSZ*NEGN)          /* 12288 latent rows */
#define OFF_LAT  0
#define OFF_WTN  (NLAT*DIM)                    /* bf16 W_node^T, ushort[128*128] */
#define OFF_WTB  (OFF_WTN + (DIM*DIM)/2)       /* bf16 W_nbr^T  */
#define OFF_POS  (OFF_WTB + (DIM*DIM)/2)       /* [2*ET*BSZ] */
#define OFF_HET  (OFF_POS + 2*ET*BSZ)
#define OFF_NEGW (OFF_HET + 2*ET*BSZ)          /* [2*ET*BSZ*NEGN] */
#define OFF_ALLP (OFF_NEGW + 2*ET*BSZ*NEGN)    /* [8] */

typedef __attribute__((ext_vector_type(16))) __bf16 v16bf;
typedef __attribute__((ext_vector_type(8)))  float  v8f;

union FragBF { v16bf v; unsigned u[8]; };

__device__ __forceinline__ unsigned short f2bf(float f) {
  union { float f; unsigned u; } x; x.f = f;
  unsigned r = x.u + 0x7FFFu + ((x.u >> 16) & 1u);   // round-to-nearest-even
  return (unsigned short)(r >> 16);
}
__device__ __forceinline__ float lk(float x)  { return x > 0.f ? x : SLOPE * x; }
__device__ __forceinline__ float sgm(float x) { return 1.f / (1.f + __expf(-x)); }

// async global -> LDS copy of 16 bytes (per-lane addresses); ASYNCcnt-tracked
__device__ __forceinline__ void async_copy_b128(unsigned lds_off, const float* gptr) {
  unsigned long long ga = (unsigned long long)(size_t)gptr;
  asm volatile("global_load_async_to_lds_b128 %0, %1, off"
               :: "v"(lds_off), "v"(ga) : "memory");
}
__device__ __forceinline__ void async_wait0() {
  asm volatile("s_wait_asynccnt 0x0" ::: "memory");
}

// A fragment: 16x32 bf16, M = lane&15; VGPR i holds K-pair ((i&4)?16:0)+g*8+(i&3)*2
__device__ __forceinline__ void loadA(FragBF& f, const unsigned short* row, int kbase, int g) {
  #pragma unroll
  for (int i = 0; i < 8; ++i) {
    int K = kbase + ((i & 4) ? 16 : 0) + g * 8 + ((i & 3) << 1);
    f.u[i] = *reinterpret_cast<const unsigned*>(row + K);
  }
}
// B fragment: 32x16 bf16, N = lane&15; VGPR i holds K-pair ((i&4)?16:0)+g*8+(i&3)*2 (+kbase)
__device__ __forceinline__ void loadB(FragBF& f, const unsigned short* row, int kbase, int g) {
  #pragma unroll
  for (int i = 0; i < 8; ++i) {
    int K = kbase + ((i & 4) ? 16 : 0) + g * 8 + ((i & 3) << 1);
    f.u[i] = *reinterpret_cast<const unsigned*>(row + K);
  }
}

// ---------------------------------------------------------------- prep ------
__global__ __launch_bounds__(256) void prep_kernel(
    const float* __restrict__ Wn, const float* __restrict__ Wb,
    const int* __restrict__ sfl, const int* __restrict__ tfl,
    unsigned short* __restrict__ WtN, unsigned short* __restrict__ WtB,
    float* __restrict__ allp)
{
  int blk = blockIdx.x, tid = threadIdx.x;
  if (blk < 16) {
    for (int i = blk * 1024 + tid; i < (blk + 1) * 1024; i += 256) {
      int k = i >> 7, n = i & 127;
      WtN[n * DIM + k] = f2bf(Wn[i]);          // transpose: [n][k]
    }
  } else if (blk < 32) {
    int bb = blk - 16;
    for (int i = bb * 1024 + tid; i < (bb + 1) * 1024; i += 256) {
      int k = i >> 7, n = i & 127;
      WtB[n * DIM + k] = f2bf(Wb[i]);
    }
  } else {
    if (tid < 2 * ET) {
      int side = tid >> 2, e = tid & 3;
      const int* fl = side ? tfl : sfl;
      int all = 1;
      for (int b = 0; b < BSZ; ++b) all &= (fl[e * BSZ + b] > 0) ? 1 : 0;
      allp[tid] = (float)all;
    }
  }
}

// --------------------------------------------------------------- latent -----
// rows: [0,B)=s_ids, [B,2B)=t_ids, [2B,2B+B*NEG)=s_negs, then t_negs
__global__ __launch_bounds__(256) void latent_kernel(
    const int* __restrict__ s_ids, const int* __restrict__ t_ids,
    const int* __restrict__ s_negs, const int* __restrict__ t_negs,
    const float* __restrict__ emb, const unsigned short* __restrict__ WtN,
    const float* __restrict__ bias, float* __restrict__ lat)
{
  __shared__ __align__(16) float Fst[16][DIM];   // 8 KB fp32 async staging
  __shared__ unsigned short Ash[16][130];        // bf16 A tile (+pad)
  __shared__ int rid[16];
  int tid = threadIdx.x;
  int rowbase = blockIdx.x * 16;

  if (tid < 16) {
    int row = rowbase + tid, id;
    if (row < BSZ)                          id = s_ids[row];
    else if (row < 2 * BSZ)                 id = t_ids[row - BSZ];
    else if (row < 2 * BSZ + BSZ * NEGN)    id = s_negs[row - 2 * BSZ];
    else                                    id = t_negs[row - 2 * BSZ - BSZ * NEGN];
    rid[tid] = id;
  }
  __syncthreads();

  // async gather: 16 rows x 32 chunks of 16B, streamed straight into LDS
  for (int ch = tid; ch < 16 * 32; ch += 256) {
    int r = ch >> 5, cc = ch & 31;
    async_copy_b128((unsigned)(size_t)&Fst[r][cc * 4],
                    emb + (long long)rid[r] * DIM + cc * 4);
  }
  async_wait0();
  __syncthreads();
  for (int i = tid; i < 16 * DIM; i += 256) {
    int r = i >> 7, c = i & 127;
    Ash[r][c] = f2bf(Fst[r][c]);
  }
  __syncthreads();

  int lane = tid & 31, wave = tid >> 5;
  int nloc = lane & 15, g = lane >> 4;
  int N = wave * 16 + nloc;                 // 8 waves cover N = 0..127
  FragBF b0, b1, b2, b3;
  loadB(b0, WtN + N * DIM, 0,  g);
  loadB(b1, WtN + N * DIM, 32, g);
  loadB(b2, WtN + N * DIM, 64, g);
  loadB(b3, WtN + N * DIM, 96, g);
  float bv = bias[N];
  v8f acc;
  #pragma unroll
  for (int r = 0; r < 8; ++r) acc[r] = bv;

  FragBF af;
  loadA(af, &Ash[nloc][0], 0,  g);
  acc = __builtin_amdgcn_wmma_f32_16x16x32_bf16(false, af.v, false, b0.v, (short)0, acc, false, false);
  loadA(af, &Ash[nloc][0], 32, g);
  acc = __builtin_amdgcn_wmma_f32_16x16x32_bf16(false, af.v, false, b1.v, (short)0, acc, false, false);
  loadA(af, &Ash[nloc][0], 64, g);
  acc = __builtin_amdgcn_wmma_f32_16x16x32_bf16(false, af.v, false, b2.v, (short)0, acc, false, false);
  loadA(af, &Ash[nloc][0], 96, g);
  acc = __builtin_amdgcn_wmma_f32_16x16x32_bf16(false, af.v, false, b3.v, (short)0, acc, false, false);

  #pragma unroll
  for (int r = 0; r < 8; ++r) {
    int M = r + 8 * g;
    lat[(long long)(rowbase + M) * DIM + N] = lk(acc[r]);
  }
}

// ----------------------------------------------------------------- nbr ------
// one block per (side, edge-type, batch element); fused latent + reductions.
// BufA aliases: phase 1 = fp32 async staging (stride 128); phase 3 = fp32
// neighbor latents (stride 129, bank-conflict padded).
__global__ __launch_bounds__(256) void nbr_kernel(
    const int* __restrict__ s_nbr_ids, const int* __restrict__ t_nbr_ids,
    const int* __restrict__ s_nbr_msk, const int* __restrict__ t_nbr_msk,
    const float* __restrict__ s_nbr_w,  const float* __restrict__ t_nbr_w,
    const float* __restrict__ emb, const float* __restrict__ ete,
    const unsigned short* __restrict__ WtB, const float* __restrict__ bias,
    const float* __restrict__ Watt, const float* __restrict__ batt,
    const float* __restrict__ lat,
    float* __restrict__ posO, float* __restrict__ hetO, float* __restrict__ negO)
{
  __shared__ __align__(16) float BufA[64 * 129];  // 33 KB (staging / nbr latents)
  __shared__ unsigned short Ash[64][130];         // 16.6 KB bf16 A tile
  __shared__ int rid[64];
  __shared__ float qv[DIM], tv[DIM], nv[NEGN][DIM];
  __shared__ float dA[64], muA[64], mgA[NEGN][64], at2[64], nwA[64];
  __shared__ float av[DIM];
  __shared__ float red[8];
#define FSTV(r, c) BufA[(r) * DIM + (c)]
#define NBRS(r, c) BufA[(r) * 129 + (c)]

  int tid  = threadIdx.x;
  int idx  = blockIdx.x;
  int side = idx / (ET * BSZ);
  int rem  = idx - side * ET * BSZ;
  int e = rem / BSZ, b = rem - e * BSZ;

  const int*   ids = (side ? t_nbr_ids : s_nbr_ids) + (e * BSZ + b) * NBRN;
  const int*   msk = (side ? t_nbr_msk : s_nbr_msk) + (e * BSZ + b) * NBRN;
  const float* wgt = (side ? t_nbr_w   : s_nbr_w)   + (e * BSZ + b) * NBRN;
  int nodeRow = side ? (BSZ + b) : b;                               // t_e : s_e
  int tgtRow  = side ? b : (BSZ + b);                               // s_e : t_e
  int negBase = side ? (2 * BSZ + b * NEGN)                         // neg_s
                     : (2 * BSZ + BSZ * NEGN + b * NEGN);           // neg_t

  if (tid < 64) rid[tid] = ids[(tid < NBRN) ? tid : 0];
  if (tid < NBRN) __builtin_prefetch(emb + (long long)ids[tid] * DIM, 0, 0);
  __syncthreads();

  // async gather: 50 rows x 32 chunks of 16B -> LDS staging
  for (int ch = tid; ch < NBRN * 32; ch += 256) {
    int r = ch >> 5, cc = ch & 31;
    async_copy_b128((unsigned)(size_t)&FSTV(r, cc * 4),
                    emb + (long long)rid[r] * DIM + cc * 4);
  }
  // overlap: pull latent vectors while the async DMA streams rows
  for (int c = tid; c < DIM; c += 256) {
    qv[c] = lat[(long long)nodeRow * DIM + c] + ete[e * DIM + c];
    tv[c] = lat[(long long)tgtRow * DIM + c];
  }
  for (int i = tid; i < NEGN * DIM; i += 256)
    nv[i >> 7][i & 127] = lat[(long long)(negBase + (i >> 7)) * DIM + (i & 127)];
  async_wait0();
  __syncthreads();

  // convert fp32 staging -> bf16 A tile (zero-pad rows >= NBRN)
  for (int i = tid; i < 64 * DIM; i += 256) {
    int r = i >> 7, c = i & 127;
    float v = (r < NBRN) ? FSTV(r, c) : 0.f;
    Ash[r][c] = f2bf(v);
  }
  __syncthreads();   // staging dead; BufA reused as nbr latents below

  int lane = tid & 31, wave = tid >> 5;
  int nloc = lane & 15, g = lane >> 4;
  int N = wave * 16 + nloc;
  FragBF b0, b1, b2, b3;
  loadB(b0, WtB + N * DIM, 0,  g);
  loadB(b1, WtB + N * DIM, 32, g);
  loadB(b2, WtB + N * DIM, 64, g);
  loadB(b3, WtB + N * DIM, 96, g);
  float bv = bias[N];
  #pragma unroll
  for (int mt = 0; mt < 4; ++mt) {
    v8f acc;
    #pragma unroll
    for (int r = 0; r < 8; ++r) acc[r] = bv;
    FragBF af;
    loadA(af, &Ash[mt * 16 + nloc][0], 0,  g);
    acc = __builtin_amdgcn_wmma_f32_16x16x32_bf16(false, af.v, false, b0.v, (short)0, acc, false, false);
    loadA(af, &Ash[mt * 16 + nloc][0], 32, g);
    acc = __builtin_amdgcn_wmma_f32_16x16x32_bf16(false, af.v, false, b1.v, (short)0, acc, false, false);
    loadA(af, &Ash[mt * 16 + nloc][0], 64, g);
    acc = __builtin_amdgcn_wmma_f32_16x16x32_bf16(false, af.v, false, b2.v, (short)0, acc, false, false);
    loadA(af, &Ash[mt * 16 + nloc][0], 96, g);
    acc = __builtin_amdgcn_wmma_f32_16x16x32_bf16(false, af.v, false, b3.v, (short)0, acc, false, false);
    #pragma unroll
    for (int r = 0; r < 8; ++r)
      NBRS(mt * 16 + r + 8 * g, N) = lk(acc[r]);
  }
  __syncthreads();

  // per-neighbor distances (7 dots over D); padded stride avoids bank conflicts
  if (tid < 64) {
    int n = tid;
    float d = 0, mu = 0, m0 = 0, m1 = 0, m2 = 0, m3 = 0, m4 = 0;
    if (n < NBRN) {
      for (int c = 0; c < DIM; ++c) {
        float x = NBRS(n, c), z;
        z = qv[c] - x;    d  += z * z;
        z = x - tv[c];    mu += z * z;
        z = x - nv[0][c]; m0 += z * z;
        z = x - nv[1][c]; m1 += z * z;
        z = x - nv[2][c]; m2 += z * z;
        z = x - nv[3][c]; m3 += z * z;
        z = x - nv[4][c]; m4 += z * z;
      }
    }
    dA[n]  = (n < NBRN && msk[n] != 0) ? -d : NEG_INF;
    muA[n] = -mu;
    mgA[0][n] = -m0; mgA[1][n] = -m1; mgA[2][n] = -m2; mgA[3][n] = -m3; mgA[4][n] = -m4;
  }
  __syncthreads();
  if (tid == 0) {
    float mx = NEG_INF;
    for (int n = 0; n < NBRN; ++n) mx = fmaxf(mx, dA[n]);
    float s = 0.f;
    for (int n = 0; n < NBRN; ++n) s += __expf(dA[n] - mx);
    red[0] = mx; red[1] = s;
  }
  __syncthreads();
  if (tid < 64) {
    float a2 = 0.f, nw = 0.f;
    if (tid < NBRN) {
      float a = __expf(dA[tid] - red[0]) / red[1];
      a2 = (msk[tid] != 0) ? a : 0.f;
      nw = a2 * wgt[tid];
    }
    at2[tid] = a2; nwA[tid] = nw;
  }
  __syncthreads();
  if (tid < DIM) {
    float s = 0.f;
    for (int n = 0; n < NBRN; ++n) s += at2[n] * NBRS(n, tid);
    av[tid] = s;                                  // avg_embed[c]
  }
  __syncthreads();
  if (tid == 0) {
    float pos = 0, n0 = 0, n1 = 0, n2 = 0, n3 = 0, n4 = 0, wsum = 0, cnt = 0;
    for (int n = 0; n < NBRN; ++n) {
      float nw = nwA[n];
      pos += nw * muA[n];
      n0 += nw * mgA[0][n]; n1 += nw * mgA[1][n]; n2 += nw * mgA[2][n];
      n3 += nw * mgA[3][n]; n4 += nw * mgA[4][n];
      wsum += wgt[n];
      cnt  += (msk[n] != 0) ? 1.f : 0.f;
    }
    cnt = fminf(fmaxf(cnt, 1.f), (float)NBRN);
    float avew = wsum / cnt;
    float h = batt[0];
    for (int c = 0; c < DIM; ++c) h += avew * av[c] * Watt[c];
    h = lk(h);
    int o = (side * ET + e) * BSZ + b;
    posO[o] = pos; hetO[o] = h;
    negO[o * NEGN + 0] = n0; negO[o * NEGN + 1] = n1; negO[o * NEGN + 2] = n2;
    negO[o * NEGN + 3] = n3; negO[o * NEGN + 4] = n4;
  }
#undef FSTV
#undef NBRS
}

// ---------------------------------------------------------------- loss ------
__global__ __launch_bounds__(128) void loss_kernel(
    const int* __restrict__ e_types,
    const int* __restrict__ sfl, const int* __restrict__ tfl,
    const float* __restrict__ ete, const float* __restrict__ lat,
    const float* __restrict__ posO, const float* __restrict__ hetO,
    const float* __restrict__ negO, const float* __restrict__ allp,
    float* __restrict__ out)
{
  __shared__ float red[DIM];
  __shared__ float red5[NEGN][DIM];
  __shared__ float bc[12];
  int b = blockIdx.x, tid = threadIdx.x;
  int et = e_types[b];
  float se = lat[(long long)b * DIM + tid];
  float te = lat[(long long)(BSZ + b) * DIM + tid];
  float dm = se + ete[et * DIM + tid] - te;
  red[tid] = dm * dm;
  #pragma unroll
  for (int j = 0; j < NEGN; ++j) {                         // s_e vs neg_t
    float d = se - lat[(long long)(2 * BSZ + BSZ * NEGN + b * NEGN + j) * DIM + tid];
    red5[j][tid] = d * d;
  }
  __syncthreads();
  if (tid == 0) {
    float s = 0; for (int c = 0; c < DIM; ++c) s += red[c];
    bc[0] = -s;
    for (int j = 0; j < NEGN; ++j) {
      float t = 0; for (int c = 0; c < DIM; ++c) t += red5[j][c];
      bc[1 + j] = -t;
    }
  }
  __syncthreads();
  #pragma unroll
  for (int j = 0; j < NEGN; ++j) {                         // t_e vs neg_s
    float d = te - lat[(long long)(2 * BSZ + b * NEGN + j) * DIM + tid];
    red5[j][tid] = d * d;
  }
  __syncthreads();
  if (tid == 0) {
    for (int j = 0; j < NEGN; ++j) {
      float t = 0; for (int c = 0; c < DIM; ++c) t += red5[j][c];
      bc[6 + j] = -t;
    }
    float posl[2], negl[2][NEGN];
    for (int side = 0; side < 2; ++side) {
      const int* fl = side ? tfl : sfl;
      float pv[ET], hv[ET], ngv[ET][NEGN], att[ET], x[ET];
      bool mk[ET];
      for (int e = 0; e < ET; ++e) {
        bool ap = allp[side * ET + e] > 0.5f;
        int o = (side * ET + e) * BSZ + b;
        pv[e] = ap ? 0.f : posO[o];
        hv[e] = ap ? 0.f : hetO[o];
        for (int j = 0; j < NEGN; ++j) ngv[e][j] = ap ? 0.f : negO[o * NEGN + j];
        mk[e] = fl[e * BSZ + b] > 0;
      }
      for (int e = 0; e < ET; ++e) att[e] = mk[e] ? hv[e] : NEG_INF;
      float mx = att[0]; for (int e = 1; e < ET; ++e) mx = fmaxf(mx, att[e]);
      float s = 0; for (int e = 0; e < ET; ++e) { att[e] = __expf(att[e] - mx); s += att[e]; }
      for (int e = 0; e < ET; ++e) att[e] /= s;            // att_v1
      for (int e = 0; e < ET; ++e) x[e] = mk[e] ? att[e] : 0.f;
      mx = x[0]; for (int e = 1; e < ET; ++e) mx = fmaxf(mx, x[e]);
      s = 0; for (int e = 0; e < ET; ++e) { x[e] = __expf(x[e] - mx); s += x[e]; }
      float pl = 0, nl[NEGN] = {0, 0, 0, 0, 0};
      for (int e = 0; e < ET; ++e) {
        float na = x[e] / s;                               // norm_att
        pl += na * pv[e];
        for (int j = 0; j < NEGN; ++j) nl[j] += na * ngv[e][j];
      }
      posl[side] = pl;
      for (int j = 0; j < NEGN; ++j) negl[side][j] = nl[j];
    }
    float lam = bc[0] + posl[0] + posl[1];
    float acc = -__logf(sgm(lam) + 1e-6f) / (float)BSZ;
    for (int j = 0; j < NEGN; ++j) {
      acc += -__logf(sgm(-(bc[1 + j] + negl[0][j])) + 1e-6f) / (float)(BSZ * NEGN);
      acc += -__logf(sgm(-(bc[6 + j] + negl[1][j])) + 1e-6f) / (float)(BSZ * NEGN);
    }
    if (b == 0) {
      float r = 0;
      for (int i = 0; i < ET * DIM; ++i) { float v = ete[i]; r += v * v; }
      acc += 1e-4f * r;
    }
    atomicAdd(out, acc);
  }
}

// ---------------------------------------------------------------- launch ----
extern "C" void kernel_launch(void* const* d_in, const int* in_sizes, int n_in,
                              void* d_out, int out_size, void* d_ws, size_t ws_size,
                              hipStream_t stream)
{
  const int*   e_types   = (const int*)  d_in[0];
  const int*   s_ids     = (const int*)  d_in[1];
  const int*   s_negs    = (const int*)  d_in[3];
  const int*   s_nbr_ids = (const int*)  d_in[4];
  const int*   s_nbr_msk = (const int*)  d_in[5];
  const float* s_nbr_w   = (const float*)d_in[6];
  const int*   s_nbr_fl  = (const int*)  d_in[7];
  const int*   t_ids     = (const int*)  d_in[8];
  const int*   t_negs    = (const int*)  d_in[10];
  const int*   t_nbr_ids = (const int*)  d_in[11];
  const int*   t_nbr_msk = (const int*)  d_in[12];
  const float* t_nbr_w   = (const float*)d_in[13];
  const int*   t_nbr_fl  = (const int*)  d_in[14];
  const float* emb       = (const float*)d_in[15];
  const float* ete       = (const float*)d_in[16];
  const float* Wn        = (const float*)d_in[17];
  const float* bn        = (const float*)d_in[18];
  const float* Wb        = (const float*)d_in[19];
  const float* bb        = (const float*)d_in[20];
  const float* Wa        = (const float*)d_in[21];
  const float* ba        = (const float*)d_in[22];

  float* ws  = (float*)d_ws;
  float* lat = ws + OFF_LAT;
  unsigned short* WtN = (unsigned short*)(ws + OFF_WTN);
  unsigned short* WtB = (unsigned short*)(ws + OFF_WTB);
  float* posO = ws + OFF_POS;
  float* hetO = ws + OFF_HET;
  float* negO = ws + OFF_NEGW;
  float* allp = ws + OFF_ALLP;

  hipMemsetAsync(d_out, 0, sizeof(float), stream);
  prep_kernel<<<33, 256, 0, stream>>>(Wn, Wb, s_nbr_fl, t_nbr_fl, WtN, WtB, allp);
  latent_kernel<<<NLAT / 16, 256, 0, stream>>>(s_ids, t_ids, s_negs, t_negs, emb, WtN, bn, lat);
  nbr_kernel<<<2 * ET * BSZ, 256, 0, stream>>>(s_nbr_ids, t_nbr_ids, s_nbr_msk, t_nbr_msk,
                                               s_nbr_w, t_nbr_w, emb, ete, WtB, bb, Wa, ba,
                                               lat, posO, hetO, negO);
  loss_kernel<<<BSZ, 128, 0, stream>>>(e_types, s_nbr_fl, t_nbr_fl, ete, lat,
                                       posO, hetO, negO, allp, (float*)d_out);
}
